// DecoderModel_6751688589852
// MI455X (gfx1250) — matrix operand order repaired and checked by reading, not compiled, not run on hardware
//
#include <hip/hip_runtime.h>

#define DEV __device__ __forceinline__

typedef __bf16 bf16;
typedef __attribute__((ext_vector_type(16))) __bf16 v16bf;
typedef __attribute__((ext_vector_type(8)))  __bf16 v8bf;
typedef __attribute__((ext_vector_type(8)))  float  v8f;

static constexpr int Bz  = 64;     // batch
static constexpr int TIN = 128;    // input length
static constexpr int TOUT= 32;     // output length
static constexpr int VV  = 32000;  // vocab
static constexpr int E   = 512;    // embedding dim
static constexpr int H   = 1024;   // hidden
static constexpr int AD  = 512;    // attention dim
static constexpr long HB = 64L * 1024L;  // B*H

DEV bf16 f2bf(float f) {
    union { float f; unsigned u; } in; in.f = f;
    unsigned r = (in.u + 0x7FFFu + ((in.u >> 16) & 1u)) >> 16;
    union { unsigned short s; bf16 b; } out; out.s = (unsigned short)r;
    return out.b;
}
DEV float sigm(float x) { return 1.0f / (1.0f + __expf(-x)); }

// ---------------------------------------------------------------------------
// f32 -> bf16 conversion (grid-stride)
// ---------------------------------------------------------------------------
__global__ void k_f2bf(const float* __restrict__ s, bf16* __restrict__ d, long n) {
    long i  = (long)blockIdx.x * blockDim.x + threadIdx.x;
    long st = (long)gridDim.x * blockDim.x;
    for (; i < n; i += st) d[i] = f2bf(s[i]);
}

__global__ void k_bsum(const float* __restrict__ a, const float* __restrict__ b,
                       float* __restrict__ o, int n) {
    int i = blockIdx.x * blockDim.x + threadIdx.x;
    if (i < n) o[i] = a[i] + b[i];
}

// ---------------------------------------------------------------------------
// init: c-state copy, bf16 h slots in the xh buffers, out_rnn0 -> cat, tokens
// grid: 1024 x 256  (= 4*64*1024 threads)
// ---------------------------------------------------------------------------
__global__ __launch_bounds__(256)
void k_init(const float* __restrict__ cell0, const float* __restrict__ hidden0,
            const float* __restrict__ out0,  const int* __restrict__ eos,
            float* __restrict__ c, bf16* xh0, bf16* xh1, bf16* xh2, bf16* xh3,
            bf16* cat, int* tok) {
    int i = blockIdx.x * 256 + threadIdx.x;       // 0 .. 262143
    c[i] = cell0[i];
    int l = i >> 16, b = (i >> 10) & 63, j = i & 1023;
    bf16 hb = f2bf(hidden0[i]);
    if      (l == 0) xh0[b * 3584 + 2560 + j] = hb;
    else if (l == 1) xh1[b * 2048 + 1024 + j] = hb;
    else if (l == 2) xh2[b * 2048 + 1024 + j] = hb;
    else             xh3[b * 2048 + 1024 + j] = hb;
    if (l == 0) cat[b * 3072 + j] = f2bf(out0[b * 1024 + j]);
    if (i < Bz) tok[i] = eos[0];
}

// ---------------------------------------------------------------------------
// WMMA bf16 GEMM:  C[MxN] f32 = A[MxK]bf16 @ B[KxN]bf16 (+bias) (opt ReLU)
// block = 256 thr = 8 waves; block tile 64(M) x 128(N); wave tile 64x16.
// grid = (N/128, M/64). Requires K%32==0, M%64==0, N%128==0.
// A-frag layout (16-bit A 16x32): lane%16 = M row, K = 16*(h/8)+8*(lane/16)+h%8
//  -> two contiguous v8bf loads at k+8g and k+16+8g.
// B-frag layout (16-bit B 32x16): lane%16 = N col, K = 16*(h/8)+8*(lane/16)+h%8
//  -> 16 strided scalar loads (stride ldb) per lane.
// C/D layout: lane%16 = N col, M = 8*(lane/16) + vgpr r.
// All pointers strength-reduced; prefetch is branch-free (uniform s_cselect).
// ---------------------------------------------------------------------------
__global__ __launch_bounds__(256)
void k_gemm(const bf16* __restrict__ Am, int lda,
            const bf16* __restrict__ Bm, int ldb,
            const float* __restrict__ bias,
            float* __restrict__ C, int ldc,
            int K, int relu) {
    const int lane = threadIdx.x & 31;
    const int wave = threadIdx.x >> 5;
    const int n0   = blockIdx.x * 128 + wave * 16;
    const int m0   = blockIdx.y * 64;
    const int mrow = lane & 15;
    const int g    = lane >> 4;

    const size_t bstep = (size_t)32 * ldb;
    const bf16* bp  = Bm + (size_t)(8 * g) * ldb + n0 + mrow;
    const bf16* ar0 = Am + (size_t)(m0 + mrow) * lda + 8 * g;
    const bf16* ar1 = ar0 + (size_t)16 * lda;
    const bf16* ar2 = ar1 + (size_t)16 * lda;
    const bf16* ar3 = ar2 + (size_t)16 * lda;

    v8f acc0 = {}, acc1 = {}, acc2 = {}, acc3 = {};
    union AF { v16bf v; v8bf h[2]; };

    for (int kk = 0; kk < K; kk += 32) {
        // B fragment: col n0+lane%16, rows kk+8g+s and kk+16+8g+s (s<8)
        v16bf bf;
        #pragma unroll
        for (int s = 0; s < 8; ++s) { bf[s] = bp[(size_t)s * ldb]; bf[8 + s] = bp[(size_t)(16 + s) * ldb]; }
        // branch-free prefetch two K-tiles ahead (uniform clamp keeps it in-bounds)
        __builtin_prefetch(bp + ((kk + 96 <= K) ? 2 * bstep : 0), 0, 2);

        AF a0, a1, a2, a3;
        a0.h[0] = *(const v8bf*)(ar0); a0.h[1] = *(const v8bf*)(ar0 + 16);
        a1.h[0] = *(const v8bf*)(ar1); a1.h[1] = *(const v8bf*)(ar1 + 16);
        a2.h[0] = *(const v8bf*)(ar2); a2.h[1] = *(const v8bf*)(ar2 + 16);
        a3.h[0] = *(const v8bf*)(ar3); a3.h[1] = *(const v8bf*)(ar3 + 16);

        acc0 = __builtin_amdgcn_wmma_f32_16x16x32_bf16(false, a0.v, false, bf, (short)0, acc0, false, false);
        acc1 = __builtin_amdgcn_wmma_f32_16x16x32_bf16(false, a1.v, false, bf, (short)0, acc1, false, false);
        acc2 = __builtin_amdgcn_wmma_f32_16x16x32_bf16(false, a2.v, false, bf, (short)0, acc2, false, false);
        acc3 = __builtin_amdgcn_wmma_f32_16x16x32_bf16(false, a3.v, false, bf, (short)0, acc3, false, false);

        bp  += bstep;
        ar0 += 32; ar1 += 32; ar2 += 32; ar3 += 32;
    }

    const int n  = n0 + (lane & 15);
    const float bv = bias ? bias[n] : 0.0f;
    const int mb = m0 + 8 * g;
    #pragma unroll
    for (int r = 0; r < 8; ++r) {
        float v0 = acc0[r] + bv, v1 = acc1[r] + bv, v2 = acc2[r] + bv, v3 = acc3[r] + bv;
        if (relu) { v0 = fmaxf(v0, 0.f); v1 = fmaxf(v1, 0.f); v2 = fmaxf(v2, 0.f); v3 = fmaxf(v3, 0.f); }
        C[(size_t)(mb +  0 + r) * ldc + n] = v0;
        C[(size_t)(mb + 16 + r) * ldc + n] = v1;
        C[(size_t)(mb + 32 + r) * ldc + n] = v2;
        C[(size_t)(mb + 48 + r) * ldc + n] = v3;
    }
}

// ---------------------------------------------------------------------------
// embedding gather -> xh0 cols [2048, 2560)   grid 128 x 256
// ---------------------------------------------------------------------------
__global__ __launch_bounds__(256)
void k_gather_emb(const float* __restrict__ emb, const int* __restrict__ tok,
                  bf16* __restrict__ xh0) {
    int gid = blockIdx.x * 256 + threadIdx.x;      // 64*512
    int b = gid >> 9, e = gid & 511;
    xh0[b * 3584 + 2048 + e] = f2bf(emb[(size_t)tok[b] * E + e]);
}

// ---------------------------------------------------------------------------
// attention scores: score[t,b] = sum_a tanh(proj[b,a]+encp[t*B+b,a]) * A2[a]
//                               + b2 + (1-is_on)*(-1e30)     (b1 folded in encp)
// grid 1024 x 256: one wave per (t,b) row, wave32 shuffle reduction
// ---------------------------------------------------------------------------
__global__ __launch_bounds__(256)
void k_attn_score(const float* __restrict__ proj, const float* __restrict__ encp,
                  const float* __restrict__ A2v,  const float* __restrict__ b2,
                  const float* __restrict__ is_on, float* __restrict__ scores) {
    int lane = threadIdx.x & 31, wave = threadIdx.x >> 5;
    int i = blockIdx.x * 8 + wave;                 // 0 .. 8191 (= t*64+b)
    int b = i & 63;
    const float* pr = proj + b * AD;
    const float* ep = encp + (size_t)i * AD;
    float s = 0.0f;
    for (int a = lane; a < AD; a += 32)
        s += tanhf(pr[a] + ep[a]) * A2v[a];
    #pragma unroll
    for (int o = 16; o; o >>= 1) s += __shfl_down(s, o, 32);
    if (lane == 0)
        scores[i] = s + b2[0] + (1.0f - is_on[i]) * (-1e30f);
}

// softmax over t (axis 0), one block of 128 threads per batch column
__global__ __launch_bounds__(128)
void k_softmax_t(const float* __restrict__ sc, float* __restrict__ alpha) {
    __shared__ float red[128];
    int b = blockIdx.x, t = threadIdx.x;
    float v = sc[t * Bz + b];
    red[t] = v; __syncthreads();
    for (int o = 64; o; o >>= 1) { if (t < o) red[t] = fmaxf(red[t], red[t + o]); __syncthreads(); }
    float mx = red[0]; __syncthreads();
    float e = __expf(v - mx);
    red[t] = e; __syncthreads();
    for (int o = 64; o; o >>= 1) { if (t < o) red[t] += red[t + o]; __syncthreads(); }
    alpha[t * Bz + b] = e / red[0];
}

// context[b,d] = sum_t alpha[t,b] * enc[t,b,d]; write bf16 to xh0 & cat
// grid 512 x 256
__global__ __launch_bounds__(256)
void k_context(const float* __restrict__ alpha, const float* __restrict__ enc,
               bf16* __restrict__ xh0, bf16* __restrict__ cat) {
    int gid = blockIdx.x * 256 + threadIdx.x;      // 64*2048
    int b = gid >> 11, d = gid & 2047;
    float s = 0.0f;
    for (int t = 0; t < TIN; ++t)
        s += alpha[t * Bz + b] * enc[(size_t)(t * Bz + b) * 2048 + d];
    bf16 cv = f2bf(s);
    xh0[b * 3584 + d]        = cv;
    cat[b * 3072 + 1024 + d] = cv;
}

// LSTM pointwise (gate order i,f,g,o); writes bf16 h into next-layer input slot
// and own-h slot (read next step). grid 256 x 256
__global__ __launch_bounds__(256)
void k_lstm_pw(const float* __restrict__ gates, float* __restrict__ c,
               bf16* __restrict__ nxt, int ld_nxt,
               bf16* __restrict__ self, int ld_self) {
    int gid = blockIdx.x * 256 + threadIdx.x;      // 64*1024
    int b = gid >> 10, j = gid & 1023;
    const float* gb = gates + (size_t)b * 4096;
    float ig = sigm(gb[j]);
    float fg = sigm(gb[1024 + j]);
    float gg = tanhf(gb[2048 + j]);
    float og = sigm(gb[3072 + j]);
    float cn = fg * c[gid] + ig * gg;
    float hn = og * tanhf(cn);
    c[gid] = cn;
    bf16 hb = f2bf(hn);
    nxt[b * ld_nxt + j]  = hb;
    self[b * ld_self + j] = hb;
}

// softmax over V + argmax feedback; one block of 256 per batch row
__global__ __launch_bounds__(256)
void k_prob(const float* __restrict__ logits, float* __restrict__ out,
            int* __restrict__ tok) {
    __shared__ float sv[256];
    __shared__ int   si[256];
    int b = blockIdx.x, t = threadIdx.x;
    const float* lb = logits + (size_t)b * VV;
    float mx = -3.0e38f; int arg = 0;
    for (int j = t; j < VV; j += 256) { float v = lb[j]; if (v > mx) { mx = v; arg = j; } }
    sv[t] = mx; si[t] = arg; __syncthreads();
    for (int o = 128; o; o >>= 1) {
        if (t < o) {
            if (sv[t + o] > sv[t] || (sv[t + o] == sv[t] && si[t + o] < si[t])) {
                sv[t] = sv[t + o]; si[t] = si[t + o];
            }
        }
        __syncthreads();
    }
    float gmx = sv[0];
    if (t == 0) tok[b] = si[0];
    __syncthreads();
    float s = 0.0f;
    for (int j = t; j < VV; j += 256) s += __expf(lb[j] - gmx);
    sv[t] = s; __syncthreads();
    for (int o = 128; o; o >>= 1) { if (t < o) sv[t] += sv[t + o]; __syncthreads(); }
    float inv = 1.0f / sv[0];
    float* ob = out + (size_t)b * VV;
    for (int j = t; j < VV; j += 256) ob[j] = __expf(lb[j] - gmx) * inv;
}

// ---------------------------------------------------------------------------
extern "C" void kernel_launch(void* const* d_in, const int* in_sizes, int n_in,
                              void* d_out, int out_size, void* d_ws, size_t ws_size,
                              hipStream_t stream) {
    (void)in_sizes; (void)n_in; (void)out_size; (void)ws_size;
    const float* enc_f   = (const float*)d_in[0];
    const float* is_on   = (const float*)d_in[1];
    const float* emb     = (const float*)d_in[2];
    const float* A1f     = (const float*)d_in[3];
    const float* b1      = (const float*)d_in[4];
    const float* A2v     = (const float*)d_in[5];
    const float* b2      = (const float*)d_in[6];
    const float* W_ih0   = (const float*)d_in[7];
    const float* W_ihr   = (const float*)d_in[8];
    const float* W_hh    = (const float*)d_in[9];
    const float* b_ih    = (const float*)d_in[10];
    const float* b_hh    = (const float*)d_in[11];
    const float* Wl      = (const float*)d_in[12];
    const float* bl      = (const float*)d_in[13];
    const float* hidden0 = (const float*)d_in[14];
    const float* cell0   = (const float*)d_in[15];
    const float* out0    = (const float*)d_in[16];
    const int*   eos     = (const int*)d_in[17];

    char* ws = (char*)d_ws;
    size_t off = 0;
    auto take = [&](size_t bytes) -> char* {
        char* p = ws + off;
        off += (bytes + 255) & ~(size_t)255;
        return p;
    };
    bf16*  Wl_b  = (bf16*)take((size_t)3072 * VV * 2);        // 196.6 MB
    bf16*  Wcat0 = (bf16*)take((size_t)3584 * 4096 * 2);      // [W_ih0 ; W_hh0]
    bf16*  Wcat1 = (bf16*)take((size_t)2048 * 4096 * 2);
    bf16*  Wcat2 = (bf16*)take((size_t)2048 * 4096 * 2);
    bf16*  Wcat3 = (bf16*)take((size_t)2048 * 4096 * 2);
    bf16*  A1_b  = (bf16*)take((size_t)3072 * AD * 2);
    bf16*  enc_b = (bf16*)take((size_t)TIN * Bz * 2048 * 2);  // 33.5 MB
    float* encp  = (float*)take((size_t)TIN * Bz * AD * 4);   // enc_proj (+b1)
    float* bsum  = (float*)take((size_t)4 * 4096 * 4);
    float* proj  = (float*)take((size_t)Bz * AD * 4);
    float* scores= (float*)take((size_t)TIN * Bz * 4);
    float* alpha = (float*)take((size_t)TIN * Bz * 4);
    bf16*  xh0   = (bf16*)take((size_t)Bz * 3584 * 2);  // [ctx | emb | h0]
    bf16*  xh1   = (bf16*)take((size_t)Bz * 2048 * 2);  // [h_below | h1]
    bf16*  xh2   = (bf16*)take((size_t)Bz * 2048 * 2);
    bf16*  xh3   = (bf16*)take((size_t)Bz * 2048 * 2);
    bf16*  cat   = (bf16*)take((size_t)Bz * 3072 * 2);  // [out | ctx]
    float* gates = (float*)take((size_t)Bz * 4096 * 4);
    float* cst   = (float*)take((size_t)4 * HB * 4);
    float* logits= (float*)take((size_t)Bz * VV * 4);
    int*   tok   = (int*)take((size_t)Bz * 4);

    auto conv = [&](const float* s, bf16* d, long n) {
        long g = (n + 255) / 256; if (g > 32768) g = 32768;
        k_f2bf<<<dim3((unsigned)g), dim3(256), 0, stream>>>(s, d, n);
    };

    // ---- one-time weight conversion / packing --------------------------------
    conv(Wl,    Wl_b,  (long)3072 * VV);
    conv(W_ih0, Wcat0, (long)2560 * 4096);
    conv(W_hh,  Wcat0 + (size_t)2560 * 4096, (long)1024 * 4096);
    bf16* Wcats[4] = { Wcat0, Wcat1, Wcat2, Wcat3 };
    for (int l = 1; l < 4; ++l) {
        conv(W_ihr + (size_t)(l - 1) * 1024 * 4096, Wcats[l],                       (long)1024 * 4096);
        conv(W_hh  + (size_t)l * 1024 * 4096,       Wcats[l] + (size_t)1024 * 4096, (long)1024 * 4096);
    }
    conv(A1f,   A1_b,  (long)3072 * AD);
    conv(enc_f, enc_b, (long)TIN * Bz * 2048);
    k_bsum<<<dim3(64), dim3(256), 0, stream>>>(b_ih, b_hh, bsum, 4 * 4096);
    k_init<<<dim3(1024), dim3(256), 0, stream>>>(cell0, hidden0, out0, eos, cst,
                                                 xh0, xh1, xh2, xh3, cat, tok);

    // enc_proj = enc @ A1[H:] + b1   (M=8192, N=512, K=2048)
    k_gemm<<<dim3(AD / 128, (TIN * Bz) / 64), dim3(256), 0, stream>>>(
        enc_b, 2048, A1_b + (size_t)H * AD, AD, b1, encp, AD, 2048, 0);

    // ---- decode loop ---------------------------------------------------------
    for (int s = 0; s < TOUT; ++s) {
        // proj = out_prev @ A1[:H]   (out_prev lives in cat cols [0,1024))
        k_gemm<<<dim3(4, 1), dim3(256), 0, stream>>>(
            cat, 3072, A1_b, AD, nullptr, proj, AD, 1024, 0);
        k_gather_emb<<<dim3(128), dim3(256), 0, stream>>>(emb, tok, xh0);
        k_attn_score<<<dim3(1024), dim3(256), 0, stream>>>(proj, encp, A2v, b2, is_on, scores);
        k_softmax_t<<<dim3(Bz), dim3(128), 0, stream>>>(scores, alpha);
        k_context<<<dim3(512), dim3(256), 0, stream>>>(alpha, enc_f, xh0, cat);

        // LSTM stack: gates = [x,h] @ [W_ih;W_hh] + (b_ih+b_hh)
        k_gemm<<<dim3(32, 1), dim3(256), 0, stream>>>(xh0, 3584, Wcat0, 4096, bsum,          gates, 4096, 3584, 0);
        k_lstm_pw<<<dim3(256), dim3(256), 0, stream>>>(gates, cst,          xh1,        2048, xh0 + 2560, 3584);
        k_gemm<<<dim3(32, 1), dim3(256), 0, stream>>>(xh1, 2048, Wcat1, 4096, bsum + 4096,   gates, 4096, 2048, 0);
        k_lstm_pw<<<dim3(256), dim3(256), 0, stream>>>(gates, cst + HB,     xh2,        2048, xh1 + 1024, 2048);
        k_gemm<<<dim3(32, 1), dim3(256), 0, stream>>>(xh2, 2048, Wcat2, 4096, bsum + 8192,   gates, 4096, 2048, 0);
        k_lstm_pw<<<dim3(256), dim3(256), 0, stream>>>(gates, cst + 2 * HB, xh3,        2048, xh2 + 1024, 2048);
        k_gemm<<<dim3(32, 1), dim3(256), 0, stream>>>(xh3, 2048, Wcat3, 4096, bsum + 12288,  gates, 4096, 2048, 0);
        k_lstm_pw<<<dim3(256), dim3(256), 0, stream>>>(gates, cst + 3 * HB, cat,        3072, xh3 + 1024, 2048);

        // logits = relu([out,ctx] @ Wl + bl)   (M=64, N=32000, K=3072)
        k_gemm<<<dim3(VV / 128, 1), dim3(256), 0, stream>>>(
            cat, 3072, Wl_b, VV, bl, logits, VV, 3072, 1);

        k_prob<<<dim3(Bz), dim3(256), 0, stream>>>(
            logits, (float*)d_out + (size_t)s * Bz * VV, tok);
    }
}
